// BasicBlock_13426067767949
// MI455X (gfx1250) — 8-bit matrix operand order repaired and checked by reading, compiled, not run on hardware
//
#include <hip/hip_runtime.h>
#include <stdint.h>

typedef __attribute__((ext_vector_type(8))) int v8i;
typedef __attribute__((ext_vector_type(4))) int v4i;

#define BATCH 32
#define CIN   384
#define Hh    56
#define Ww    56
#define HW    3136
#define PADW  58
#define NPIX  (BATCH*HW)
#define NCTILE 24
#define NCOG   12
#define NPG    (NPIX/64)
#define NTILES (NCOG*NPG)
#define EPSBN 1e-5f

__device__ inline v8i load_v8(const int8_t* p) {
    const v4i* q = (const v4i*)p;
    v4i lo = q[0], hi = q[1];
    v8i r;
    r[0]=lo[0]; r[1]=lo[1]; r[2]=lo[2]; r[3]=lo[3];
    r[4]=hi[0]; r[5]=hi[1]; r[6]=hi[2]; r[7]=hi[3];
    return r;
}

__global__ void zero_kernel(int* p, long n) {
    for (long i = blockIdx.x*(long)blockDim.x + threadIdx.x; i < n;
         i += (long)gridDim.x*blockDim.x) p[i] = 0;
}

__global__ void pack_weights(const float* __restrict__ w, int8_t* __restrict__ wp, long total) {
    for (long t = blockIdx.x*(long)blockDim.x + threadIdx.x; t < total;
         t += (long)gridDim.x*blockDim.x) {
        int f    = (int)(t >> 10);
        int r    = (int)(t & 1023);
        int lane = r >> 5;
        int q    = r & 31;
        int v    = q >> 2;
        int bb   = q & 3;
        int chunk = f % 6; int f2 = f / 6;
        int tap   = f2 % 9; int cot = f2 / 9;
        int kl = ((lane&16)?32:0) + q;
        int co = cot*16 + (lane & 15);
        int ci = chunk*64 + kl;
        int dy = tap/3, dx = tap%3;
        float val = w[(((long)co*CIN + ci)*3 + dy)*3 + dx];
        wp[t] = (int8_t)((val > 0.f) - (val < 0.f));
    }
}

__global__ void binarize_x(const float* __restrict__ x, int8_t* __restrict__ sx) {
    const long n = (long)BATCH*CIN*HW;
    for (long i = blockIdx.x*(long)blockDim.x + threadIdx.x; i < n;
         i += (long)gridDim.x*blockDim.x) {
        int c  = (int)((i / HW) % CIN);
        int b  = (int)(i / ((long)CIN*HW));
        int p  = (int)(i % HW);
        int y  = p / Ww, xx = p % Ww;
        float v = x[i];
        sx[(((long)(b*PADW + y + 1))*PADW + (xx + 1))*CIN + c] =
            (int8_t)((v > 0.f) - (v < 0.f));
    }
}

template<int MODE>
__global__ __launch_bounds__(256)
void conv_wmma(const int8_t* __restrict__ sx, const int8_t* __restrict__ wp,
               const float* __restrict__ resid, float* __restrict__ out) {
    const int lane = threadIdx.x & 31;
    const int wave = threadIdx.x >> 5;
    const int tile = blockIdx.x*8 + wave;
    const int cog  = tile % NCOG;
    const int ng   = tile / NCOG;
    const int khalf = (lane & 16) ? 32 : 0;

    int  pb[4], pp[4];
    long pixbase[4];
    #pragma unroll
    for (int j = 0; j < 4; ++j) {
        int pix = ng*64 + j*16 + (lane & 15);
        pb[j] = pix / HW;
        pp[j] = pix % HW;
        int y = pp[j] / Ww, xx = pp[j] % Ww;
        pixbase[j] = (((long)(pb[j]*PADW + y))*PADW + xx)*CIN + khalf;
    }
    const int8_t* wbase0 = wp + (long)((cog*2    )*9)*6*1024 + lane*32;
    const int8_t* wbase1 = wp + (long)((cog*2 + 1)*9)*6*1024 + lane*32;

    v8i acc[2][4];
    #pragma unroll
    for (int i = 0; i < 2; ++i)
        #pragma unroll
        for (int j = 0; j < 4; ++j) acc[i][j] = (v8i){};

    #pragma unroll 1
    for (int tap = 0; tap < 9; ++tap) {
        const long toff = (long)((tap/3)*PADW + (tap%3))*CIN;
        if (tap < 8) {
            const long noff = (long)(((tap+1)/3)*PADW + ((tap+1)%3))*CIN;
            #pragma unroll
            for (int j = 0; j < 4; ++j)
                __builtin_prefetch(sx + pixbase[j] + noff, 0, 1);
        }
        const int8_t* wt0 = wbase0 + tap*6*1024;
        const int8_t* wt1 = wbase1 + tap*6*1024;
        #pragma unroll 1
        for (int ch = 0; ch < 6; ++ch) {
            v8i A0 = load_v8(wt0 + ch*1024);
            v8i A1 = load_v8(wt1 + ch*1024);
            v8i B0 = load_v8(sx + pixbase[0] + toff + ch*64);
            v8i B1 = load_v8(sx + pixbase[1] + toff + ch*64);
            v8i B2 = load_v8(sx + pixbase[2] + toff + ch*64);
            v8i B3 = load_v8(sx + pixbase[3] + toff + ch*64);
            acc[0][0] = __builtin_amdgcn_wmma_i32_16x16x64_iu8(true, A0, true, B0, acc[0][0], false, false);
            acc[0][1] = __builtin_amdgcn_wmma_i32_16x16x64_iu8(true, A0, true, B1, acc[0][1], false, false);
            acc[0][2] = __builtin_amdgcn_wmma_i32_16x16x64_iu8(true, A0, true, B2, acc[0][2], false, false);
            acc[0][3] = __builtin_amdgcn_wmma_i32_16x16x64_iu8(true, A0, true, B3, acc[0][3], false, false);
            acc[1][0] = __builtin_amdgcn_wmma_i32_16x16x64_iu8(true, A1, true, B0, acc[1][0], false, false);
            acc[1][1] = __builtin_amdgcn_wmma_i32_16x16x64_iu8(true, A1, true, B1, acc[1][1], false, false);
            acc[1][2] = __builtin_amdgcn_wmma_i32_16x16x64_iu8(true, A1, true, B2, acc[1][2], false, false);
            acc[1][3] = __builtin_amdgcn_wmma_i32_16x16x64_iu8(true, A1, true, B3, acc[1][3], false, false);
        }
    }

    const int chh = (lane & 16) ? 8 : 0;
    #pragma unroll
    for (int i = 0; i < 2; ++i) {
        const int chbase = cog*32 + i*16 + chh;
        #pragma unroll
        for (int j = 0; j < 4; ++j) {
            if (MODE == 1) {
                long pix = (long)pb[j]*HW + pp[j];
                float* o = out + pix*CIN + chbase;
                #pragma unroll
                for (int r = 0; r < 8; ++r) o[r] = (float)acc[i][j][r];
            } else {
                #pragma unroll
                for (int r = 0; r < 8; ++r) {
                    long idx = ((long)pb[j]*CIN + (chbase + r))*HW + pp[j];
                    out[idx] = (float)acc[i][j][r] + resid[idx];
                }
            }
        }
    }
}

__global__ __launch_bounds__(256)
void stats_nhwc(const float* __restrict__ yv, float* __restrict__ sum,
                float* __restrict__ ssum, long n) {
    __shared__ float ls[CIN], lss[CIN];
    for (int i = threadIdx.x; i < CIN; i += blockDim.x) { ls[i] = 0.f; lss[i] = 0.f; }
    __syncthreads();
    for (long i = blockIdx.x*(long)blockDim.x + threadIdx.x; i < n;
         i += (long)gridDim.x*blockDim.x) {
        float v = yv[i];
        int ch = (int)(i % CIN);
        atomicAdd(&ls[ch], v);
        atomicAdd(&lss[ch], v*v);
    }
    __syncthreads();
    for (int i = threadIdx.x; i < CIN; i += blockDim.x) {
        atomicAdd(&sum[i], ls[i]);
        atomicAdd(&ssum[i], lss[i]);
    }
}

__global__ __launch_bounds__(256)
void stats_nchw(const float* __restrict__ z, float* __restrict__ sum,
                float* __restrict__ ssum) {
    __shared__ float s1[256], s2[256];
    const int c    = blockIdx.x % CIN;
    const int slab = blockIdx.x / CIN;
    float a = 0.f, b = 0.f;
    for (int bb = slab; bb < BATCH; bb += 8) {
        const float* base = z + ((long)bb*CIN + c)*HW;
        for (int i = threadIdx.x; i < HW; i += 256) { float v = base[i]; a += v; b += v*v; }
    }
    s1[threadIdx.x] = a; s2[threadIdx.x] = b;
    __syncthreads();
    for (int off = 128; off > 0; off >>= 1) {
        if (threadIdx.x < off) {
            s1[threadIdx.x] += s1[threadIdx.x + off];
            s2[threadIdx.x] += s2[threadIdx.x + off];
        }
        __syncthreads();
    }
    if (threadIdx.x == 0) { atomicAdd(&sum[c], s1[0]); atomicAdd(&ssum[c], s2[0]); }
}

__global__ void bn_scale(const float* __restrict__ sum, const float* __restrict__ ssum,
                         const float* __restrict__ g, const float* __restrict__ beta,
                         float* __restrict__ a, float* __restrict__ c) {
    int i = blockIdx.x*blockDim.x + threadIdx.x;
    if (i < CIN) {
        const float inv_n = 1.0f / (float)NPIX;
        float m   = sum[i]*inv_n;
        float var = ssum[i]*inv_n - m*m;
        float sc  = g[i] * rsqrtf(var + EPSBN);
        a[i] = sc;
        c[i] = beta[i] - sc*m;
    }
}

__global__ void binarize_bn(const float* __restrict__ yv, const float* __restrict__ a,
                            const float* __restrict__ c, int8_t* __restrict__ sx) {
    const long n = (long)NPIX*CIN;
    for (long i = blockIdx.x*(long)blockDim.x + threadIdx.x; i < n;
         i += (long)gridDim.x*blockDim.x) {
        int  ch  = (int)(i % CIN);
        long pix = i / CIN;
        int  b   = (int)(pix / HW);
        int  p   = (int)(pix % HW);
        int  y   = p / Ww, xx = p % Ww;
        float v  = a[ch]*yv[i] + c[ch];
        sx[(((long)(b*PADW + y + 1))*PADW + (xx + 1))*CIN + ch] =
            (int8_t)((v > 0.f) - (v < 0.f));
    }
}

__global__ void bn_clip_out(const float* __restrict__ z, const float* __restrict__ a,
                            const float* __restrict__ c, float* __restrict__ o) {
    const long n = (long)BATCH*CIN*HW;
    for (long i = blockIdx.x*(long)blockDim.x + threadIdx.x; i < n;
         i += (long)gridDim.x*blockDim.x) {
        int ch  = (int)((i / HW) % CIN);
        float v = a[ch]*z[i] + c[ch];
        o[i] = fminf(1.0f, fmaxf(-1.0f, v));
    }
}

extern "C" void kernel_launch(void* const* d_in, const int* in_sizes, int n_in,
                              void* d_out, int out_size, void* d_ws, size_t ws_size,
                              hipStream_t stream) {
    const float* x  = (const float*)d_in[0];
    const float* W1 = (const float*)d_in[1];
    const float* W2 = (const float*)d_in[2];
    const float* g1 = (const float*)d_in[3];
    const float* b1 = (const float*)d_in[4];
    const float* g2 = (const float*)d_in[5];
    const float* b2 = (const float*)d_in[6];
    float* out = (float*)d_out;

    float*  zbuf = (float*)d_ws;
    size_t  zoff = (size_t)NPIX*CIN*sizeof(float);
    int8_t* sx   = (int8_t*)d_ws + zoff;
    const size_t sxsize = (size_t)BATCH*PADW*PADW*CIN;
    int8_t* w1p  = sx + sxsize;
    const long wpbytes = (long)NCTILE*9*6*1024;
    int8_t* w2p  = w1p + wpbytes;
    float*  st   = (float*)(w2p + wpbytes);
    float *sum1 = st,          *ss1 = st + CIN,   *a1 = st + 2*CIN, *c1 = st + 3*CIN;
    float *sum2 = st + 4*CIN,  *ss2 = st + 5*CIN, *a2 = st + 6*CIN, *c2 = st + 7*CIN;

    zero_kernel<<<4096, 256, 0, stream>>>((int*)sx, (long)(sxsize/4));
    zero_kernel<<<4,    256, 0, stream>>>((int*)st, 8*CIN);

    pack_weights<<<(int)((wpbytes + 255)/256), 256, 0, stream>>>(W1, w1p, wpbytes);
    pack_weights<<<(int)((wpbytes + 255)/256), 256, 0, stream>>>(W2, w2p, wpbytes);

    binarize_x<<<8192, 256, 0, stream>>>(x, sx);

    conv_wmma<1><<<NTILES/8, 256, 0, stream>>>(sx, w1p, nullptr, zbuf);

    stats_nhwc<<<4096, 256, 0, stream>>>(zbuf, sum1, ss1, (long)NPIX*CIN);
    bn_scale<<<2, 256, 0, stream>>>(sum1, ss1, g1, b1, a1, c1);

    binarize_bn<<<8192, 256, 0, stream>>>(zbuf, a1, c1, sx);

    conv_wmma<2><<<NTILES/8, 256, 0, stream>>>(sx, w2p, x, zbuf);

    stats_nchw<<<CIN*8, 256, 0, stream>>>(zbuf, sum2, ss2);
    bn_scale<<<2, 256, 0, stream>>>(sum2, ss2, g2, b2, a2, c2);
    bn_clip_out<<<8192, 256, 0, stream>>>(zbuf, a2, c2, out);
}
